// EquivariantVAE_28089086116410
// MI455X (gfx1250) — matrix operand order repaired and checked by reading, not compiled, not run on hardware
//
#include <hip/hip_runtime.h>
#include <hip/hip_bf16.h>

typedef __attribute__((ext_vector_type(16))) __bf16       v16bf;
typedef __attribute__((ext_vector_type(8)))  float        v8f;
typedef __attribute__((ext_vector_type(4)))  unsigned int u32x4;
typedef __attribute__((ext_vector_type(4)))  float        f32x4;

#define NW 58
constexpr long BATCH    = 16384;
constexpr long OUT_CVAE = BATCH * 1600;   // c_vae_out after v_raw
constexpr long OUT_MU   = BATCH * 3200;   // mu
constexpr long OUT_LV   = BATCH * 3600;   // logvar_expanded

// LDS element offsets (bf16/ushort units)
#define XOFF 0        // 25600: INP layout (also reused as HID buffer)
#define YOFF 25600    // 17408: HID layout
#define ZOFF 43008    // 12800: LAT layout (Kpad=32, zero-padded)
#define GOFF 55808    // 16*192 floats = 6144 ushort slots
#define SMLEN 61952   // 123904 bytes total

// ---------------- weight block table (58 blocks, d_in[1+i]) ----------------
struct WTab { int K[NW]; int N[NW]; unsigned off[NW]; unsigned total; };
constexpr WTab make_wtab() {
  WTab t{};
  const int k[NW] = {
    64,64,64,64,64,64,          // enc_lin1
    64,64,64,64,32,32,          // enc_lin2
    64,64,64,64,32,32,          // enc_lin3
    64,64,64,32,32,             // enc_mu
    64,                         // enc_logvar
    16,16,16,16,16,16,          // dec_lin1
    64,64,64,64,32,32,          // dec_lin2
    64,64,64,64,32,32,          // dec_lin3
    64,64,64,32,32,             // dec_out
    64,64,64,64,32,32,          // recon_lin1
    64,64,64,32,32 };           // recon_out
  const int n[NW] = {
    64,192,64,64,32,32,
    64,192,64,64,32,32,
    64,192,64,64,32,32,
    16,16,16,16,16,
    80,
    64,192,64,64,32,32,
    64,192,64,64,32,32,
    64,192,64,64,32,32,
    64,64,64,64,64,
    64,192,64,64,32,32,
    64,64,64,64,64 };
  unsigned o = 0;
  for (int i = 0; i < NW; ++i) {
    t.K[i] = k[i]; t.N[i] = n[i]; t.off[i] = o;
    unsigned kp = (unsigned)((k[i] + 31) & ~31);
    o += (kp / 32u) * (unsigned)(n[i] / 16) * 512u;   // 512 bf16 per 32x16 fragment
  }
  t.total = o;
  return t;
}
__constant__ WTab WT = make_wtab();
constexpr  WTab HWT = make_wtab();

// ---------------- job tables ----------------
struct GJobs { unsigned char l[64], mm[64], nt[64]; };
constexpr GJobs make_gjobs() {
  GJobs g{}; const int nn[5] = {0,4,4,2,2}; const int ml[5] = {1,3,5,7,9}; int idx = 0;
  for (int l = 1; l < 5; ++l)
    for (int mm = 0; mm < ml[l]; ++mm)
      for (int nt = 0; nt < nn[l]; ++nt) { g.l[idx]=(unsigned char)l; g.mm[idx]=(unsigned char)mm; g.nt[idx]=(unsigned char)nt; ++idx; }
  return g;
}
__constant__ GJobs GJ = make_gjobs();

struct MJobs { unsigned char l[25], mm[25]; };
constexpr MJobs make_mjobs() {
  MJobs g{}; const int ml[5] = {1,3,5,7,9}; int idx = 0;
  for (int l = 0; l < 5; ++l)
    for (int mm = 0; mm < ml[l]; ++mm) { g.l[idx]=(unsigned char)l; g.mm[idx]=(unsigned char)mm; ++idx; }
  return g;
}
__constant__ MJobs MJ = make_mjobs();

struct OJobs { unsigned char l[100], mm[100], nt[100]; };
constexpr OJobs make_ojobs() {
  OJobs g{}; const int ml[5] = {1,3,5,7,9}; int idx = 0;
  for (int l = 0; l < 5; ++l)
    for (int mm = 0; mm < ml[l]; ++mm)
      for (int nt = 0; nt < 4; ++nt) { g.l[idx]=(unsigned char)l; g.mm[idx]=(unsigned char)mm; g.nt[idx]=(unsigned char)nt; ++idx; }
  return g;
}
__constant__ OJobs OJ = make_ojobs();

// layouts: 0=INP, 1=HID, 2=LAT.  IB = per-l base offset, KPt = per-l padded K
__constant__ int IB[3][5]  = { {0,1024,4096,9216,16384}, {0,1024,4096,9216,12800}, {0,512,2048,4608,8192} };
__constant__ int KPt[3][5] = { {64,64,64,64,64}, {64,64,64,32,32}, {32,32,32,32,32} };
__constant__ int MLc[5]   = {1,3,5,7,9};
__constant__ int MOFF[5]  = {0,16,64,144,256};     // e3nn offsets for 16-mul irreps
__constant__ int VOFF[5]  = {0,64,256,576,1024};   // e3nn offsets for 64-mul irreps
__constant__ int CROW[5]  = {0,1,4,9,16};          // row offsets in (25,64) c layout
__constant__ int GOFFL[5] = {0,0,64,128,160};      // gate channel offsets per l
__constant__ int NNG[5]   = {0,4,4,2,2};           // N/16 of gated out blocks

// ---------------- helpers ----------------
__device__ __forceinline__ unsigned short f2bf(float f) {
  unsigned u = __float_as_uint(f);
  u += 0x7FFFu + ((u >> 16) & 1u);            // round-to-nearest-even
  return (unsigned short)(u >> 16);
}
__device__ __forceinline__ float sigm(float x) { return 1.0f / (1.0f + __expf(-x)); }

union FragAB { v16bf v; u32x4 q[2]; };

// C += A(16xKpad bf16, LDS) * B(Kpad x 16 bf16, pre-fragmented in ws)
__device__ __forceinline__ v8f gemm16(const unsigned short* sm, int inBase, int kpad, int mm,
                                      const unsigned short* ws, int widx, int nn, int nt, int lane) {
  const u32x4* wb = (const u32x4*)(ws + WT.off[widx]);
  const int half = lane >> 4, tb = lane & 15;
  const int rowBase = inBase + (mm * 16 + tb) * kpad;
  v8f acc = {};
  const int nkt = kpad >> 5;
  for (int kt = 0; kt < nkt; ++kt) {
    FragAB a, b;
    // A layout (16-bit 16x32): e0..7 -> K = half*8+e ; e8..15 -> K = 16+half*8+e
    int o = rowBase + kt * 32 + half * 8;
    a.q[0] = *(const u32x4*)(sm + o);
    a.q[1] = *(const u32x4*)(sm + o + 16);
    // B fragment pre-swizzled: [kt][nt][lane][16] contiguous
    const u32x4* p = wb + ((size_t)kt * nn + nt) * 64 + lane * 2;
    b.q[0] = p[0];
    b.q[1] = p[1];
    acc = __builtin_amdgcn_wmma_f32_16x16x32_bf16(false, a.v, false, b.v, (short)0, acc, false, false);
  }
  return acc;
}

// one e3nn Linear(PRE)+Gate layer: in (layout li @ inOff) -> out (HID layout @ outOff)
__device__ __forceinline__ void gated_layer(unsigned short* sm, int li, int inOff, int outOff,
                                            float* gates, const unsigned short* ws, int w0,
                                            int lane, int wave) {
  const int col = lane & 15, half = lane >> 4;
  // phase A: l=0 block, 256 cols = 64 silu scalars + 192 sigmoid gates
  for (int job = wave; job < 16; job += 8) {
    int widx, ntl, nn;
    if (job < 4) { widx = w0;     ntl = job;     nn = 4;  }
    else         { widx = w0 + 1; ntl = job - 4; nn = 12; }
    v8f acc = gemm16(sm, inOff + IB[li][0], KPt[li][0], 0, ws, widx, nn, ntl, lane);
    const int j = job * 16 + col;
#pragma unroll
    for (int r = 0; r < 8; ++r) {
      int tb = r + half * 8;
      float v = acc[r];
      if (j < 64) sm[outOff + tb * 64 + j] = f2bf(v * sigm(v));     // silu -> l=0 out
      else        gates[tb * 192 + (j - 64)] = sigm(v);             // gates buffer
    }
  }
  __syncthreads();
  // phase B: gated l>0 blocks
  for (int job = wave; job < 64; job += 8) {
    const int l = GJ.l[job], mm = GJ.mm[job], nt = GJ.nt[job];
    v8f acc = gemm16(sm, inOff + IB[li][l], KPt[li][l], mm, ws, w0 + 1 + l, NNG[l], nt, lane);
    const int ch  = nt * 16 + col;
    const int go  = GOFFL[l] + ch;
    const int okp = KPt[1][l];
    const int ob  = outOff + IB[1][l] + ch;
#pragma unroll
    for (int r = 0; r < 8; ++r) {
      int tb = r + half * 8;
      sm[ob + (mm * 16 + tb) * okp] = f2bf(acc[r] * gates[tb * 192 + go]);
    }
  }
  __syncthreads();
}

// ---------------- kernels ----------------
struct WPtrs { const float* p[NW]; };

// fp32 weights -> bf16 WMMA B-fragments in ws (1/sqrt(fan_in) folded, K zero-padded to 32)
__global__ __launch_bounds__(256) void prep_weights(WPtrs wp, unsigned short* __restrict__ ws) {
  const int bi = blockIdx.x;
  const int K = WT.K[bi], N = WT.N[bi];
  const unsigned base = WT.off[bi];
  const int kpad = (K + 31) & ~31;
  const int nn = N >> 4, nkt = kpad >> 5;
  const int total = nkt * nn * 512;
  const float scale = rsqrtf((float)K);
  const float* W = wp.p[bi];
  for (int f = threadIdx.x; f < total; f += 256) {
    int frag = f >> 9, within = f & 511;
    int lane = within >> 4, e = within & 15;
    int kt = frag / nn, nt = frag - kt * nn;
    int c = lane & 15, h = lane >> 4;
    int k = kt * 32 + (e < 8 ? h * 8 + e : 16 + h * 8 + (e - 8));   // B layout: e0..7 -> K = half*8+e ; e8..15 -> K = 16+half*8+(e-8)
    float v = (k < K) ? W[(size_t)k * N + nt * 16 + c] * scale : 0.0f;
    ws[base + f] = f2bf(v);
  }
}

__global__ __launch_bounds__(256) void vae_main(const float* __restrict__ cin,
                                                const unsigned short* __restrict__ ws,
                                                float* __restrict__ out) {
  __shared__ __attribute__((aligned(16))) unsigned short sm[SMLEN];
  float* gates = (float*)(sm + GOFF);
  const int tid = threadIdx.x, lane = tid & 31, wave = tid >> 5;
  const int col = lane & 15, half = lane >> 4;
  const long bbase = (long)blockIdx.x * 16;

  // stage input tile: c (B,25,64) fp32 -> X bf16 as [lrow][tb][64] (c_to_e3nn is implicit)
  const float* cptr = cin + bbase * 1600;
  for (int f4 = tid; f4 < 6400; f4 += 256) {
    int f = f4 * 4;
    int lrow = f >> 10, rem = f & 1023;
    int tb = rem >> 6, i = rem & 63;
    f32x4 v = *(const f32x4*)(cptr + (long)tb * 1600 + lrow * 64 + i);
    sm[XOFF + f + 0] = f2bf(v.x); sm[XOFF + f + 1] = f2bf(v.y);
    sm[XOFF + f + 2] = f2bf(v.z); sm[XOFF + f + 3] = f2bf(v.w);
  }
  __syncthreads();

  gated_layer(sm, 0, XOFF, YOFF, gates, ws, 0,  lane, wave);   // enc_lin1  X->Y
  gated_layer(sm, 1, YOFF, XOFF, gates, ws, 6,  lane, wave);   // enc_lin2  Y->X
  gated_layer(sm, 1, XOFF, YOFF, gates, ws, 12, lane, wave);   // enc_lin3  X->Y

  // mu (-> Z LDS + global) and logvar (-> expanded global), both read Y
  for (int job = wave; job < 30; job += 8) {
    if (job < 25) {
      const int l = MJ.l[job], mm = MJ.mm[job];
      v8f acc = gemm16(sm, YOFF + IB[1][l], KPt[1][l], mm, ws, 18 + l, 1, 0, lane);
      const int m = MLc[l];
#pragma unroll
      for (int r = 0; r < 8; ++r) {
        int tb = r + half * 8;
        float v = acc[r];
        int zi = ZOFF + IB[2][l] + (mm * 16 + tb) * 32 + col;
        sm[zi]      = f2bf(v);     // z = mu (eval mode)
        sm[zi + 16] = 0;           // zero-pad K 16..31
        out[OUT_MU + (bbase + tb) * 400 + MOFF[l] + col * m + mm] = v;
      }
    } else {
      const int nt = job - 25;     // each 16-col tile == one l
      v8f acc = gemm16(sm, YOFF + IB[1][0], KPt[1][0], 0, ws, 23, 5, nt, lane);
      const int m = MLc[nt];
#pragma unroll
      for (int r = 0; r < 8; ++r) {
        int tb = r + half * 8;
        float v = acc[r];
        long b0 = OUT_LV + (bbase + tb) * 400 + MOFF[nt] + (long)col * m;
        for (int c = 0; c < m; ++c) out[b0 + c] = v;   // expand_scalars
      }
    }
  }
  __syncthreads();

  gated_layer(sm, 2, ZOFF, XOFF, gates, ws, 24, lane, wave);   // dec_lin1  Z->X
  gated_layer(sm, 1, XOFF, YOFF, gates, ws, 30, lane, wave);   // dec_lin2  X->Y
  gated_layer(sm, 1, YOFF, XOFF, gates, ws, 36, lane, wave);   // dec_lin3  Y->X (h)

  // dec_out: h(X) -> v_raw (e3nn layout, global fp32)
  for (int job = wave; job < 100; job += 8) {
    const int l = OJ.l[job], mm = OJ.mm[job], nt = OJ.nt[job];
    v8f acc = gemm16(sm, XOFF + IB[1][l], KPt[1][l], mm, ws, 42 + l, 4, nt, lane);
    const int m = MLc[l], ch = nt * 16 + col;
#pragma unroll
    for (int r = 0; r < 8; ++r) {
      long row = bbase + r + half * 8;
      out[row * 1600 + VOFF[l] + (long)ch * m + mm] = acc[r];
    }
  }
  __syncthreads();

  gated_layer(sm, 1, XOFF, YOFF, gates, ws, 47, lane, wave);   // recon_lin1 X->Y

  // recon_out: r(Y) -> c_vae_out ((25,64) c layout, global fp32)
  for (int job = wave; job < 100; job += 8) {
    const int l = OJ.l[job], mm = OJ.mm[job], nt = OJ.nt[job];
    v8f acc = gemm16(sm, YOFF + IB[1][l], KPt[1][l], mm, ws, 53 + l, 4, nt, lane);
    const int ch = nt * 16 + col;
#pragma unroll
    for (int r = 0; r < 8; ++r) {
      long row = bbase + r + half * 8;
      out[OUT_CVAE + row * 1600 + (CROW[l] + mm) * 64 + ch] = acc[r];
    }
  }
}

extern "C" void kernel_launch(void* const* d_in, const int* in_sizes, int n_in,
                              void* d_out, int out_size, void* d_ws, size_t ws_size,
                              hipStream_t stream) {
  (void)in_sizes; (void)out_size;
  if (n_in < 1 + NW) return;
  if (ws_size < (size_t)HWT.total * 2) return;
  WPtrs wp;
  for (int i = 0; i < NW; ++i) wp.p[i] = (const float*)d_in[1 + i];
  unsigned short* ws = (unsigned short*)d_ws;
  prep_weights<<<NW, 256, 0, stream>>>(wp, ws);
  vae_main<<<(int)(BATCH / 16), 256, 0, stream>>>((const float*)d_in[0], ws, (float*)d_out);
}
